// MultiHeadAttentionV2_78984448573888
// MI455X (gfx1250) — matrix operand order repaired and checked by reading, not compiled, not run on hardware
//
#include <hip/hip_runtime.h>

#define D_MODEL 1024
#define NUM_HEADS 16
#define D_HEAD 64
#define BATCH 4
#define SEQ 2048
#define ROWS (BATCH * SEQ)   // 8192

typedef __attribute__((ext_vector_type(16))) __bf16 bf16x16;
typedef __attribute__((ext_vector_type(8)))  float  f32x8;

union Frag {
    bf16x16 v;
    uint4 u[2];
    unsigned short s[16];
};

__device__ __forceinline__ unsigned short f2bf(float f) {
    unsigned u = __builtin_bit_cast(unsigned, f);
    unsigned r = u + 0x7FFFu + ((u >> 16) & 1u);   // round-to-nearest-even
    return (unsigned short)(r >> 16);
}

__device__ __forceinline__ f32x8 wmma_bf16(const Frag& a, const Frag& b, f32x8 c) {
    return __builtin_amdgcn_wmma_f32_16x16x32_bf16(
        /*neg_a=*/false, a.v, /*neg_b=*/false, b.v,
        /*c_mod=*/(short)0, c, /*reuse_a=*/false, /*reuse_b=*/false);
}

// ---------------- Kernel 1: x fp32 -> bf16 ----------------
__global__ void k_cvt_x(const float* __restrict__ x, unsigned short* __restrict__ xb, int n) {
    int i = blockIdx.x * blockDim.x + threadIdx.x;
    if (i < n) xb[i] = f2bf(x[i]);
}

// ---------------- Kernel 2: W fp32 -> bf16, transposed (Wt[n][k] = W[k][n]) ----------------
__global__ void k_cvt_w(const float* __restrict__ Wq, const float* __restrict__ Wk,
                        const float* __restrict__ Wv, unsigned short* __restrict__ Wt3) {
    const float* W = (blockIdx.y == 0) ? Wq : (blockIdx.y == 1) ? Wk : Wv;
    unsigned short* o = Wt3 + (size_t)blockIdx.y * D_MODEL * D_MODEL;
    int gid = blockIdx.x * blockDim.x + threadIdx.x;   // 0 .. 1M-1
    int k = gid >> 10;
    int n = gid & (D_MODEL - 1);
    o[(size_t)n * D_MODEL + k] = f2bf(W[(size_t)gid]);
}

// ---------------- Kernel 3: QKV projection GEMM (one 16x64 tile per wave; A reused x4) ----------------
// which = blockIdx.y: 0 -> Q (row-major bf16), 1 -> K (row-major bf16), 2 -> V (stored Vt[b][h][d][s])
__global__ __launch_bounds__(256) void k_qkv_gemm(
    const unsigned short* __restrict__ Xb,
    const unsigned short* __restrict__ Wt3,
    unsigned short* __restrict__ Qb,
    unsigned short* __restrict__ Kb,
    unsigned short* __restrict__ Vt) {
    const int which = blockIdx.y;
    const unsigned short* Wt = Wt3 + (size_t)which * D_MODEL * D_MODEL;

    const int wave = (blockIdx.x * blockDim.x + threadIdx.x) >> 5;  // 0..8191
    const int lane = threadIdx.x & 31;
    const int mt = wave >> 4;        // 512 row tiles (16 rows each)
    const int nt = wave & 15;        // 16 col tiles (64 cols each)
    const int m0 = mt * 16, n0 = nt * 64;
    const int r = lane & 15, half = lane >> 4;

    f32x8 acc0 = {}, acc1 = {}, acc2 = {}, acc3 = {};
    const unsigned short* arow  = Xb + (size_t)(m0 + r) * D_MODEL;
    const unsigned short* brow0 = Wt + (size_t)(n0 + r)      * D_MODEL;
    const unsigned short* brow1 = Wt + (size_t)(n0 + 16 + r) * D_MODEL;
    const unsigned short* brow2 = Wt + (size_t)(n0 + 32 + r) * D_MODEL;
    const unsigned short* brow3 = Wt + (size_t)(n0 + 48 + r) * D_MODEL;

    #pragma unroll 2
    for (int kb = 0; kb < D_MODEL; kb += 32) {
        Frag a, b0, b1, b2, b3;
        a.u[0]  = *(const uint4*)(arow + kb + half * 16);       // A: K = half*16 + i
        a.u[1]  = *(const uint4*)(arow + kb + half * 16 + 8);
        b0.u[0] = *(const uint4*)(brow0 + kb + half * 16);      // B: K = half*16 + i
        b0.u[1] = *(const uint4*)(brow0 + kb + half * 16 + 8);
        b1.u[0] = *(const uint4*)(brow1 + kb + half * 16);
        b1.u[1] = *(const uint4*)(brow1 + kb + half * 16 + 8);
        b2.u[0] = *(const uint4*)(brow2 + kb + half * 16);
        b2.u[1] = *(const uint4*)(brow2 + kb + half * 16 + 8);
        b3.u[0] = *(const uint4*)(brow3 + kb + half * 16);
        b3.u[1] = *(const uint4*)(brow3 + kb + half * 16 + 8);
        acc0 = wmma_bf16(a, b0, acc0);
        acc1 = wmma_bf16(a, b1, acc1);
        acc2 = wmma_bf16(a, b2, acc2);
        acc3 = wmma_bf16(a, b3, acc3);
    }

    if (which < 2) {
        unsigned short* O = (which == 0) ? Qb : Kb;
        #pragma unroll
        for (int j = 0; j < 8; j++) {
            int m = m0 + j + 8 * half;
            unsigned short* orow = O + (size_t)m * D_MODEL + n0 + r;
            orow[0]  = f2bf(acc0[j]);
            orow[16] = f2bf(acc1[j]);
            orow[32] = f2bf(acc2[j]);
            orow[48] = f2bf(acc3[j]);
        }
    } else {
        #pragma unroll
        for (int j = 0; j < 8; j++) {
            int m = m0 + j + 8 * half;
            int bb = m >> 11, s = m & (SEQ - 1);
            float vals[4] = {acc0[j], acc1[j], acc2[j], acc3[j]};
            #pragma unroll
            for (int t = 0; t < 4; t++) {
                int col = n0 + t * 16 + r;
                int h = col >> 6, d = col & 63;
                Vt[(((size_t)bb * NUM_HEADS + h) * D_HEAD + d) * SEQ + s] = f2bf(vals[t]);
            }
        }
    }
}

// ---------------- Kernel 4: causal attention, flash-style, one 16-row q-tile per wave ----------------
__global__ __launch_bounds__(256) void k_attn(
    const unsigned short* __restrict__ Qb,
    const unsigned short* __restrict__ Kb,
    const unsigned short* __restrict__ Vt,
    float* __restrict__ out) {
    __shared__ __align__(16) unsigned short Plds[8][16][40];   // 16x32 P per wave, padded rows (80B)

    const int wid = threadIdx.x >> 5, lane = threadIdx.x & 31;
    const int r = lane & 15, half = lane >> 4;
    const int b = blockIdx.z, h = blockIdx.y;
    const int q0 = (blockIdx.x * 8 + wid) * 16;

    // Q A-fragments for d 0..31 and 32..63
    const unsigned short* qrow = Qb + (size_t)(b * SEQ + q0 + r) * D_MODEL + h * D_HEAD;
    Frag aq0, aq1;
    aq0.u[0] = *(const uint4*)(qrow + half * 16);
    aq0.u[1] = *(const uint4*)(qrow + half * 16 + 8);
    aq1.u[0] = *(const uint4*)(qrow + 32 + half * 16);
    aq1.u[1] = *(const uint4*)(qrow + 32 + half * 16 + 8);

    f32x8 o0 = {}, o1 = {}, o2 = {}, o3 = {};
    float M[8], L[8];
    #pragma unroll
    for (int j = 0; j < 8; j++) { M[j] = -1e30f; L[j] = 0.f; }

    const float scale = 0.125f;                   // 1/sqrt(64)
    const float log2e = 1.4426950408889634f;

    for (int kb = 0; kb <= q0 + 16 && kb < SEQ; kb += 32) {
        // ---- scores: S[16q x 32k] as two 16x16 C tiles ----
        const unsigned short* k0row = Kb + (size_t)(b * SEQ + kb + r) * D_MODEL + h * D_HEAD;
        int c1 = kb + 16 + r; if (c1 > SEQ - 1) c1 = SEQ - 1;   // clamp OOB key row (masked below)
        const unsigned short* k1row = Kb + (size_t)(b * SEQ + c1) * D_MODEL + h * D_HEAD;

        Frag b00, b01, b10, b11;
        b00.u[0] = *(const uint4*)(k0row + half * 16);
        b00.u[1] = *(const uint4*)(k0row + half * 16 + 8);
        b01.u[0] = *(const uint4*)(k0row + 32 + half * 16);
        b01.u[1] = *(const uint4*)(k0row + 32 + half * 16 + 8);
        b10.u[0] = *(const uint4*)(k1row + half * 16);
        b10.u[1] = *(const uint4*)(k1row + half * 16 + 8);
        b11.u[0] = *(const uint4*)(k1row + 32 + half * 16);
        b11.u[1] = *(const uint4*)(k1row + 32 + half * 16 + 8);

        f32x8 s0 = {}, s1 = {};
        s0 = wmma_bf16(aq0, b00, s0);
        s0 = wmma_bf16(aq1, b01, s0);
        s1 = wmma_bf16(aq0, b10, s1);
        s1 = wmma_bf16(aq1, b11, s1);

        // ---- mask + online softmax (row stats shared by 16 lanes of same half) ----
        float p0[8], p1[8], alpha[8];
        #pragma unroll
        for (int j = 0; j < 8; j++) {
            int row = q0 + j + 8 * half;
            float v0 = s0[j] * scale;
            float v1 = s1[j] * scale;
            int col0 = kb + r;
            int col1 = kb + 16 + r;
            if (col0 > row + 1)                v0 = -1e30f;
            if (col1 > row + 1 || col1 >= SEQ) v1 = -1e30f;
            float t = fmaxf(v0, v1);
            t = fmaxf(t, __shfl_xor(t, 1));
            t = fmaxf(t, __shfl_xor(t, 2));
            t = fmaxf(t, __shfl_xor(t, 4));
            t = fmaxf(t, __shfl_xor(t, 8));
            float Mn = fmaxf(M[j], t);
            float a  = exp2f((M[j] - Mn) * log2e);
            float e0 = exp2f((v0 - Mn) * log2e);
            float e1 = exp2f((v1 - Mn) * log2e);
            float rs = e0 + e1;
            rs += __shfl_xor(rs, 1);
            rs += __shfl_xor(rs, 2);
            rs += __shfl_xor(rs, 4);
            rs += __shfl_xor(rs, 8);
            L[j] = L[j] * a + rs;
            M[j] = Mn;
            alpha[j] = a;
            p0[j] = e0; p1[j] = e1;
        }
        #pragma unroll
        for (int j = 0; j < 8; j++) {
            o0[j] *= alpha[j]; o1[j] *= alpha[j];
            o2[j] *= alpha[j]; o3[j] *= alpha[j];
        }

        // ---- P: C-layout -> LDS -> A-layout (per-wave private region; LDS is in-order) ----
        #pragma unroll
        for (int j = 0; j < 8; j++) {
            Plds[wid][j + 8 * half][r]      = f2bf(p0[j]);
            Plds[wid][j + 8 * half][16 + r] = f2bf(p1[j]);
        }
        __builtin_amdgcn_wave_barrier();
        asm volatile("s_wait_dscnt 0" ::: "memory");
        Frag ap;
        ap.u[0] = *(const uint4*)(&Plds[wid][r][half * 16]);
        ap.u[1] = *(const uint4*)(&Plds[wid][r][half * 16 + 8]);

        // ---- O += P @ V : one K=32 WMMA per 16-wide d tile ----
        int vs0 = kb + half * 16;
        if (vs0 > SEQ - 16) vs0 = SEQ - 16;    // clamped rows multiply P==0
        const size_t vbase = ((size_t)(b * NUM_HEADS + h)) * D_HEAD * SEQ;
        {
            Frag bv; const unsigned short* vr = Vt + vbase + (size_t)(0 * 16 + r) * SEQ;
            bv.u[0] = *(const uint4*)(vr + vs0); bv.u[1] = *(const uint4*)(vr + vs0 + 8);
            o0 = wmma_bf16(ap, bv, o0);
        }
        {
            Frag bv; const unsigned short* vr = Vt + vbase + (size_t)(1 * 16 + r) * SEQ;
            bv.u[0] = *(const uint4*)(vr + vs0); bv.u[1] = *(const uint4*)(vr + vs0 + 8);
            o1 = wmma_bf16(ap, bv, o1);
        }
        {
            Frag bv; const unsigned short* vr = Vt + vbase + (size_t)(2 * 16 + r) * SEQ;
            bv.u[0] = *(const uint4*)(vr + vs0); bv.u[1] = *(const uint4*)(vr + vs0 + 8);
            o2 = wmma_bf16(ap, bv, o2);
        }
        {
            Frag bv; const unsigned short* vr = Vt + vbase + (size_t)(3 * 16 + r) * SEQ;
            bv.u[0] = *(const uint4*)(vr + vs0); bv.u[1] = *(const uint4*)(vr + vs0 + 8);
            o3 = wmma_bf16(ap, bv, o3);
        }
    }

    // ---- normalize + store fp32 output [b][s][h*64+d] ----
    #pragma unroll
    for (int j = 0; j < 8; j++) {
        float inv = 1.0f / L[j];
        int row = q0 + j + 8 * half;
        float* orow = out + (size_t)(b * SEQ + row) * D_MODEL + h * D_HEAD;
        orow[0 * 16 + r] = o0[j] * inv;
        orow[1 * 16 + r] = o1[j] * inv;
        orow[2 * 16 + r] = o2[j] * inv;
        orow[3 * 16 + r] = o3[j] * inv;
    }
}

extern "C" void kernel_launch(void* const* d_in, const int* in_sizes, int n_in,
                              void* d_out, int out_size, void* d_ws, size_t ws_size,
                              hipStream_t stream) {
    (void)in_sizes; (void)n_in; (void)out_size; (void)ws_size;
    const float* x  = (const float*)d_in[0];
    const float* Wq = (const float*)d_in[1];
    const float* Wk = (const float*)d_in[2];
    const float* Wv = (const float*)d_in[3];
    float* out = (float*)d_out;

    char* ws = (char*)d_ws;
    size_t off = 0;
    unsigned short* Xb  = (unsigned short*)(ws + off); off += (size_t)ROWS * D_MODEL * 2;        // 16 MB
    unsigned short* Wt3 = (unsigned short*)(ws + off); off += (size_t)3 * D_MODEL * D_MODEL * 2; // 6 MB
    unsigned short* Qb  = (unsigned short*)(ws + off); off += (size_t)ROWS * D_MODEL * 2;        // 16 MB
    unsigned short* Kb  = (unsigned short*)(ws + off); off += (size_t)ROWS * D_MODEL * 2;        // 16 MB
    unsigned short* Vt  = (unsigned short*)(ws + off); off += (size_t)ROWS * D_MODEL * 2;        // 16 MB

    // 1) x -> bf16
    k_cvt_x<<<(ROWS * D_MODEL) / 256, 256, 0, stream>>>(x, Xb, ROWS * D_MODEL);
    // 2) W -> bf16 transposed (3 weights via grid.y)
    k_cvt_w<<<dim3((D_MODEL * D_MODEL) / 256, 3), 256, 0, stream>>>(Wq, Wk, Wv, Wt3);
    // 3) QKV projections: 8192 waves (16x64 tiles), 8 waves/block; grid.y selects Q/K/V
    k_qkv_gemm<<<dim3(1024, 3), 256, 0, stream>>>(Xb, Wt3, Qb, Kb, Vt);
    // 4) attention: 128 q-tiles per (b,h), 8 waves/block
    k_attn<<<dim3(SEQ / 16 / 8, NUM_HEADS, BATCH), 256, 0, stream>>>(Qb, Kb, Vt, out);
}
